// DiffEncoder_79482664780191
// MI455X (gfx1250) — matrix operand order repaired and checked by reading, not compiled, not run on hardware
//
#include <hip/hip_runtime.h>
#include <hip/hip_bf16.h>
#include <math.h>

// ---------------------------------------------------------------------------
// Types for CDNA5 WMMA (wave32, v_wmma_f32_16x16x32_bf16)
// ---------------------------------------------------------------------------
typedef __bf16 bf16;
typedef __attribute__((ext_vector_type(16))) __bf16 v16bf;
typedef __attribute__((ext_vector_type(8)))  __bf16 v8bf;
typedef __attribute__((ext_vector_type(8)))  float  v8f;
typedef __attribute__((ext_vector_type(4)))  int    v4i;

// gfx1250 async global->LDS copy (ASYNCcnt-tracked), guarded by __has_builtin
#if defined(__AMDGCN__) && __has_builtin(__builtin_amdgcn_global_load_async_to_lds_b128) && __has_builtin(__builtin_amdgcn_s_wait_asynccnt)
#define USE_ASYNC_LDS 1
#else
#define USE_ASYNC_LDS 0
#endif

// Problem constants (from reference)
#define Bc    64
#define Nc    256
#define XDc   118
#define EDc   5
#define Hc    384
#define NHc   8
#define DHc   48
#define DFFc  1536
#define DEPTHc 6
#define KCP   1408   // embed K (1398) zero-padded to multiple of 32 / 16B alignment

#define AS_STR 40    // LDS row stride (elements) for A tile (128x32)
#define BS_STR 40    // LDS row stride (elements) for B tile stored [n][k] (64x32)

// ---------------------------------------------------------------------------
// Generic batched bf16 GEMM, f32 accumulate via WMMA.
//   out = alpha * A(MxK) @ op(B)(KxN) + bias[n] + addmat[m][n]; optional ReLU.
// Workgroup: 256 threads = 8 wave32; wave computes 32x32 (2x2 WMMA tiles);
// workgroup tile 128x64; K pipelined 32 at a time through double-buffered LDS,
// one barrier per K-step. Fast paths either async-copy global->LDS (gfx1250
// GLOBAL_LOAD_ASYNC_TO_LDS_B128, ASYNCcnt) or register-staged b128 copies.
// ---------------------------------------------------------------------------
template<bool TRANSB, bool RELU>
__global__ __launch_bounds__(256) void gemm_bf16_kernel(
    const bf16* __restrict__ A, const bf16* __restrict__ Bm,
    const float* __restrict__ bias, const float* __restrict__ addmat,
    float* __restrict__ outF, bf16* __restrict__ outB,
    int M, int N, int K, int lda, int ldb, int ldo, int ldadd,
    long long sA1, long long sA2, long long sB1, long long sB2,
    long long sO1, long long sO2, long long sAd1, long long sAd2,
    int nb2, float alpha)
{
    __shared__ bf16 As[2][128 * AS_STR];
    __shared__ bf16 Bs[2][64 * BS_STR];

    const int bz = blockIdx.z;
    const int b1 = bz / nb2, b2 = bz % nb2;
    A  += (long long)b1 * sA1 + (long long)b2 * sA2;
    Bm += (long long)b1 * sB1 + (long long)b2 * sB2;
    const long long oOff = (long long)b1 * sO1 + (long long)b2 * sO2;
    if (addmat) addmat += (long long)b1 * sAd1 + (long long)b2 * sAd2;

    const int m0 = blockIdx.y * 128;
    const int n0 = blockIdx.x * 64;
    const int tid  = threadIdx.x;
    const int lane = tid & 31;
    const int wave = tid >> 5;          // 0..7
    const int wm = (wave >> 1) * 32;    // 0,32,64,96
    const int wn = (wave & 1) * 32;     // 0,32
    const int lr = lane & 15;
    const int hk = lane >> 4;

    const bool fullM = (m0 + 128 <= M);
    const bool fullN = (n0 + 64  <= N);

    // Staging chunk coordinates (8 bf16 = 16B per chunk)
    const int ar = tid >> 2, akc = (tid & 3) * 8;   // A rows 0..63 (chunk0), +64 (chunk1)
    const int bnr = tid >> 2, bnk = (tid & 3) * 8;  // TRANSB: B row n, along k
    const int btk = tid >> 3, btn = (tid & 7) * 8;  // !TRANSB: B row k, along n

    v8f acc[2][2];
#pragma unroll
    for (int i = 0; i < 2; ++i)
#pragma unroll
        for (int j = 0; j < 2; ++j) acc[i][j] = (v8f)(0.0f);

    auto computeTile = [&](int p) {
        // A 16x32: lane lr = row M; half-wave hk: K in {hk*8..+7, 16+hk*8..+7}
        // B 32x16: lane lr = col N; half-wave hk: K in {hk*8..+7, 16+hk*8..+7}
        v16bf af[2], bfr[2];
#pragma unroll
        for (int t = 0; t < 2; ++t) {
            const bf16* pa = &As[p][(wm + t * 16 + lr) * AS_STR + hk * 8];
            v8bf alo = *(const v8bf*)(pa);
            v8bf ahi = *(const v8bf*)(pa + 16);
            af[t] = __builtin_shufflevector(alo, ahi,
                     0,1,2,3,4,5,6,7,8,9,10,11,12,13,14,15);
            const bf16* pb = &Bs[p][(wn + t * 16 + lr) * BS_STR + hk * 8];
            v8bf blo = *(const v8bf*)(pb);
            v8bf bhi = *(const v8bf*)(pb + 16);
            bfr[t] = __builtin_shufflevector(blo, bhi,
                     0,1,2,3,4,5,6,7,8,9,10,11,12,13,14,15);
        }
#pragma unroll
        for (int ti = 0; ti < 2; ++ti)
#pragma unroll
            for (int tj = 0; tj < 2; ++tj)
                acc[ti][tj] = __builtin_amdgcn_wmma_f32_16x16x32_bf16(
                    false, af[ti], false, bfr[tj], (short)0, acc[ti][tj],
                    false, false);
    };

#if USE_ASYNC_LDS
    // ------------------------------------------------------------------
    // Async variant: issue GLOBAL_LOAD_ASYNC_TO_LDS_B128 per 16B chunk.
    // Builtin signature (from clang diagnostic): (v4i* src, v4i* dst, int, int)
    // ------------------------------------------------------------------
    auto issueTile = [&](int p, int k0) {
        const bool fk = (k0 + 32 <= K);
        // ---- A tile ----
        if (fullM && fk) {
            __builtin_amdgcn_global_load_async_to_lds_b128(
                (v4i*)(A + (long long)(m0 + ar) * lda + (k0 + akc)),
                (v4i*)&As[p][ar * AS_STR + akc], 0, 0);
            __builtin_amdgcn_global_load_async_to_lds_b128(
                (v4i*)(A + (long long)(m0 + ar + 64) * lda + (k0 + akc)),
                (v4i*)&As[p][(ar + 64) * AS_STR + akc], 0, 0);
        } else {
            v8bf a0, a1;
#pragma unroll
            for (int q = 0; q < 8; ++q) {
                int gk = k0 + akc + q;
                int g0 = m0 + ar, g1 = g0 + 64;
                a0[q] = (g0 < M && gk < K) ? A[(long long)g0 * lda + gk] : (bf16)0.0f;
                a1[q] = (g1 < M && gk < K) ? A[(long long)g1 * lda + gk] : (bf16)0.0f;
            }
            *(v8bf*)&As[p][ar * AS_STR + akc] = a0;
            *(v8bf*)&As[p][(ar + 64) * AS_STR + akc] = a1;
        }
        // ---- B tile ----
        if (TRANSB) {
            if (fullN && fk) {
                __builtin_amdgcn_global_load_async_to_lds_b128(
                    (v4i*)(Bm + (long long)(n0 + bnr) * ldb + (k0 + bnk)),
                    (v4i*)&Bs[p][bnr * BS_STR + bnk], 0, 0);
            } else {
                v8bf b;
                int gn = n0 + bnr;
#pragma unroll
                for (int q = 0; q < 8; ++q) {
                    int gk = k0 + bnk + q;
                    b[q] = (gn < N && gk < K) ? Bm[(long long)gn * ldb + gk] : (bf16)0.0f;
                }
                *(v8bf*)&Bs[p][bnr * BS_STR + bnk] = b;
            }
        } else {
            v8bf b;
            if (fullN && fk) {
                b = *(const v8bf*)(Bm + (long long)(k0 + btk) * ldb + (n0 + btn));
            } else {
                int gk = k0 + btk;
#pragma unroll
                for (int q = 0; q < 8; ++q) {
                    int gn = n0 + btn + q;
                    b[q] = (gn < N && gk < K) ? Bm[(long long)gk * ldb + gn] : (bf16)0.0f;
                }
            }
#pragma unroll
            for (int q = 0; q < 8; ++q)
                Bs[p][(btn + q) * BS_STR + btk] = b[q];
        }
    };

    issueTile(0, 0);
    int p = 0;
    for (int k0 = 0; k0 < K; k0 += 32) {
        __builtin_amdgcn_s_wait_asynccnt(0);
        __syncthreads();
        if (k0 + 32 < K) issueTile(p ^ 1, k0 + 32);
        computeTile(p);
        p ^= 1;
    }
#else
    // ------------------------------------------------------------------
    // Register-staged pipeline fallback (no async builtins available).
    // ------------------------------------------------------------------
    auto loadTile = [&](int k0, v8bf& a0, v8bf& a1, v8bf& b) {
        const bool fk = (k0 + 32 <= K);
        if (fullM && fk) {
            a0 = *(const v8bf*)(A + (long long)(m0 + ar) * lda + (k0 + akc));
            a1 = *(const v8bf*)(A + (long long)(m0 + ar + 64) * lda + (k0 + akc));
        } else {
#pragma unroll
            for (int q = 0; q < 8; ++q) {
                int gk = k0 + akc + q;
                int g0 = m0 + ar, g1 = g0 + 64;
                a0[q] = (g0 < M && gk < K) ? A[(long long)g0 * lda + gk] : (bf16)0.0f;
                a1[q] = (g1 < M && gk < K) ? A[(long long)g1 * lda + gk] : (bf16)0.0f;
            }
        }
        if (TRANSB) {
            if (fullN && fk) {
                b = *(const v8bf*)(Bm + (long long)(n0 + bnr) * ldb + (k0 + bnk));
            } else {
                int gn = n0 + bnr;
#pragma unroll
                for (int q = 0; q < 8; ++q) {
                    int gk = k0 + bnk + q;
                    b[q] = (gn < N && gk < K) ? Bm[(long long)gn * ldb + gk] : (bf16)0.0f;
                }
            }
        } else {
            if (fullN && fk) {
                b = *(const v8bf*)(Bm + (long long)(k0 + btk) * ldb + (n0 + btn));
            } else {
                int gk = k0 + btk;
#pragma unroll
                for (int q = 0; q < 8; ++q) {
                    int gn = n0 + btn + q;
                    b[q] = (gn < N && gk < K) ? Bm[(long long)gk * ldb + gn] : (bf16)0.0f;
                }
            }
        }
    };

    auto stageStore = [&](int p, v8bf a0, v8bf a1, v8bf b) {
        *(v8bf*)&As[p][ar * AS_STR + akc] = a0;
        *(v8bf*)&As[p][(ar + 64) * AS_STR + akc] = a1;
        if (TRANSB) {
            *(v8bf*)&Bs[p][bnr * BS_STR + bnk] = b;
        } else {
#pragma unroll
            for (int q = 0; q < 8; ++q)
                Bs[p][(btn + q) * BS_STR + btk] = b[q];
        }
    };

    v8bf ra0, ra1, rb;
    loadTile(0, ra0, ra1, rb);
    int p = 0;
    for (int k0 = 0; k0 < K; k0 += 32) {
        stageStore(p, ra0, ra1, rb);
        __syncthreads();
        if (k0 + 32 < K) {
            loadTile(k0 + 32, ra0, ra1, rb);
            if (k0 + 64 < K && fullM && fullN) {
                __builtin_prefetch(A + (long long)(m0 + ar) * lda + (k0 + 64 + akc), 0, 1);
                if (TRANSB)
                    __builtin_prefetch(Bm + (long long)(n0 + bnr) * ldb + (k0 + 64 + bnk), 0, 1);
                else
                    __builtin_prefetch(Bm + (long long)(k0 + 64 + btk) * ldb + (n0 + btn), 0, 1);
            }
        }
        computeTile(p);
        p ^= 1;
    }
#endif

    // ---- epilogue: C/D layout: VGPR r, lanes<16 -> M=r, lanes>=16 -> M=8+r; N=lane&15
    if (fullM && fullN) {
#pragma unroll
        for (int ti = 0; ti < 2; ++ti) {
#pragma unroll
            for (int tj = 0; tj < 2; ++tj) {
                int nIdx = n0 + wn + tj * 16 + lr;
                float bv = bias ? bias[nIdx] : 0.0f;
                int mB = m0 + wm + ti * 16 + hk * 8;
#pragma unroll
                for (int r = 0; r < 8; ++r) {
                    int mIdx = mB + r;
                    float v = alpha * acc[ti][tj][r] + bv;
                    if (addmat) v += addmat[(long long)mIdx * ldadd + nIdx];
                    if (RELU) v = fmaxf(v, 0.0f);
                    long long o = oOff + (long long)mIdx * ldo + nIdx;
                    if (outF) outF[o] = v;
                    if (outB) outB[o] = (bf16)v;
                }
            }
        }
    } else {
#pragma unroll
        for (int ti = 0; ti < 2; ++ti) {
#pragma unroll
            for (int tj = 0; tj < 2; ++tj) {
                int nIdx = n0 + wn + tj * 16 + lr;
                if (nIdx >= N) continue;
                float bv = bias ? bias[nIdx] : 0.0f;
                int mB = m0 + wm + ti * 16 + hk * 8;
#pragma unroll
                for (int r = 0; r < 8; ++r) {
                    int mIdx = mB + r;
                    if (mIdx < M) {
                        float v = alpha * acc[ti][tj][r] + bv;
                        if (addmat) v += addmat[(long long)mIdx * ldadd + nIdx];
                        if (RELU) v = fmaxf(v, 0.0f);
                        long long o = oOff + (long long)mIdx * ldo + nIdx;
                        if (outF) outF[o] = v;
                        if (outB) outB[o] = (bf16)v;
                    }
                }
            }
        }
    }
}

// ---------------------------------------------------------------------------
// f32 -> bf16 convert
// ---------------------------------------------------------------------------
__global__ void f2bf_kernel(const float* __restrict__ in, bf16* __restrict__ out,
                            long long n)
{
    long long i = (long long)blockIdx.x * 256 + threadIdx.x;
    if (i < n) out[i] = (bf16)in[i];
}

// ---------------------------------------------------------------------------
// Build concat([x, e.reshape(b,n,-1)]) row-major bf16: [16384 x 1408] (padded)
// ---------------------------------------------------------------------------
__global__ __launch_bounds__(256) void build_cat_kernel(const float* __restrict__ x,
                                                        const float* __restrict__ e,
                                                        bf16* __restrict__ cat)
{
    const int m = blockIdx.x;                       // 0..16383
    const float* xr = x + (long long)m * XDc;
    const float* er = e + (long long)m * (Nc * EDc);
    bf16* cr = cat + (long long)m * KCP;
    for (int k = threadIdx.x; k < KCP; k += 256) {
        float v = 0.0f;
        if (k < XDc)                 v = xr[k];
        else if (k < XDc + Nc * EDc) v = er[k - XDc];
        cr[k] = (bf16)v;
    }
}

// ---------------------------------------------------------------------------
// Distance bias from Gram matrix G[b] = z_b @ z_b^T  (per-row block, 256 thr)
// ---------------------------------------------------------------------------
__global__ __launch_bounds__(256) void dbias_kernel(const float* __restrict__ G,
                                                    const float* __restrict__ mask,
                                                    float* __restrict__ D)
{
    __shared__ float red[256];
    int row = blockIdx.x;            // b*256 + i
    int b = row >> 8, i = row & 255;
    const float* gb = G + (long long)b * Nc * Nc;
    int j = threadIdx.x;
    float gij = gb[(long long)i * Nc + j];
    float gii = gb[(long long)i * Nc + i];
    float gjj = gb[(long long)j * Nc + j];
    float mi = mask[row];
    float mj = mask[(b << 8) + j];
    float d2 = fmaxf(gii + gjj - 2.0f * gij, 0.0f);
    float d = (d2 > 1e-12f) ? sqrtf(d2) : 0.0f;
    float dm = mi * mj;
    d *= dm;
    red[j] = d; __syncthreads();
    for (int s = 128; s > 0; s >>= 1) {
        if (j < s) red[j] = fmaxf(red[j], red[j + s]);
        __syncthreads();
    }
    float dmax = red[0];
    float out = (j == i) ? 0.0f : (dmax - d) * dm;
    D[(long long)row * Nc + j] = out;
}

// ---------------------------------------------------------------------------
// Row softmax over 256 cols; writes bf16 probabilities
// ---------------------------------------------------------------------------
__global__ __launch_bounds__(256) void softmax_kernel(const float* __restrict__ S,
                                                      bf16* __restrict__ P)
{
    __shared__ float red[256];
    long long row = blockIdx.x;
    int j = threadIdx.x;
    float v = S[row * Nc + j];
    red[j] = v; __syncthreads();
    for (int s = 128; s > 0; s >>= 1) {
        if (j < s) red[j] = fmaxf(red[j], red[j + s]);
        __syncthreads();
    }
    float m = red[0]; __syncthreads();
    float e = __expf(v - m);
    red[j] = e; __syncthreads();
    for (int s = 128; s > 0; s >>= 1) {
        if (j < s) red[j] += red[j + s];
        __syncthreads();
    }
    float inv = 1.0f / red[0];
    P[row * Nc + j] = (bf16)(e * inv);
}

// ---------------------------------------------------------------------------
// Fused residual add + LayerNorm over H=384; writes f32 and bf16
// ---------------------------------------------------------------------------
__global__ __launch_bounds__(128) void addnorm_kernel(
    const float* __restrict__ a, const float* __restrict__ b,
    const float* __restrict__ g, const float* __restrict__ beta,
    float* __restrict__ oF, bf16* __restrict__ oB)
{
    __shared__ float red[128];
    long long row = blockIdx.x;
    int t = threadIdx.x;
    const float* pa = a + row * Hc;
    const float* pb = b + row * Hc;
    float v0 = pa[t]       + pb[t];
    float v1 = pa[t + 128] + pb[t + 128];
    float v2 = pa[t + 256] + pb[t + 256];
    red[t] = v0 + v1 + v2; __syncthreads();
    for (int s = 64; s > 0; s >>= 1) {
        if (t < s) red[t] += red[t + s];
        __syncthreads();
    }
    float mean = red[0] * (1.0f / (float)Hc); __syncthreads();
    float d0 = v0 - mean, d1 = v1 - mean, d2 = v2 - mean;
    red[t] = d0 * d0 + d1 * d1 + d2 * d2; __syncthreads();
    for (int s = 64; s > 0; s >>= 1) {
        if (t < s) red[t] += red[t + s];
        __syncthreads();
    }
    float rstd = rsqrtf(red[0] * (1.0f / (float)Hc) + 1e-5f);
    long long o = row * Hc;
    float r0 = d0 * rstd * g[t]       + beta[t];
    float r1 = d1 * rstd * g[t + 128] + beta[t + 128];
    float r2 = d2 * rstd * g[t + 256] + beta[t + 256];
    oF[o + t] = r0; oF[o + t + 128] = r1; oF[o + t + 256] = r2;
    oB[o + t] = (bf16)r0; oB[o + t + 128] = (bf16)r1; oB[o + t + 256] = (bf16)r2;
}

// ---------------------------------------------------------------------------
// Output masking kernels
// ---------------------------------------------------------------------------
__global__ void mask_rows_kernel(const float* __restrict__ in,
                                 const float* __restrict__ mask,
                                 float* __restrict__ out, int inner, long long total)
{
    long long i = (long long)blockIdx.x * 256 + threadIdx.x;
    if (i < total) out[i] = in[i] * mask[i / inner];
}

__global__ void mask_e_kernel(const float* __restrict__ e,
                              const float* __restrict__ mask,
                              float* __restrict__ out, long long total)
{
    long long i = (long long)blockIdx.x * 256 + threadIdx.x;
    if (i < total) {
        long long pair = i / EDc;          // (b*N + n)*N + j
        int j = (int)(pair % Nc);
        long long bn = pair / Nc;          // b*N + n
        long long b = bn / Nc;
        out[i] = e[i] * mask[bn] * mask[b * Nc + j];
    }
}

// ---------------------------------------------------------------------------
// Host-side GEMM launcher
// ---------------------------------------------------------------------------
template<bool TB, bool RELU>
static void launch_gemm(hipStream_t st,
    const bf16* A, const bf16* B, const float* bias, const float* addm,
    float* oF, bf16* oB, int M, int N, int K, int lda, int ldb, int ldo, int ldadd,
    long long sA1, long long sA2, long long sB1, long long sB2,
    long long sO1, long long sO2, long long sAd1, long long sAd2,
    int nbatch, int nb2, float alpha)
{
    dim3 grid((unsigned)((N + 63) / 64), (unsigned)((M + 127) / 128), (unsigned)nbatch);
    gemm_bf16_kernel<TB, RELU><<<grid, dim3(256), 0, st>>>(
        A, B, bias, addm, oF, oB, M, N, K, lda, ldb, ldo, ldadd,
        sA1, sA2, sB1, sB2, sO1, sO2, sAd1, sAd2, nb2, alpha);
}

// ---------------------------------------------------------------------------
// kernel_launch
// ---------------------------------------------------------------------------
extern "C" void kernel_launch(void* const* d_in, const int* in_sizes, int n_in,
                              void* d_out, int out_size, void* d_ws, size_t ws_size,
                              hipStream_t stream)
{
    (void)in_sizes; (void)n_in; (void)out_size; (void)ws_size;

    const float* x     = (const float*)d_in[0];
    const float* e     = (const float*)d_in[1];
    const float* mask  = (const float*)d_in[2];
    const float* W_emb = (const float*)d_in[3];
    const float* Wq = (const float*)d_in[4];  const float* bq = (const float*)d_in[5];
    const float* Wk = (const float*)d_in[6];  const float* bk = (const float*)d_in[7];
    const float* Wv = (const float*)d_in[8];  const float* bv = (const float*)d_in[9];
    const float* Wo = (const float*)d_in[10]; const float* bo = (const float*)d_in[11];
    const float* ln1g = (const float*)d_in[12]; const float* ln1b = (const float*)d_in[13];
    const float* W1 = (const float*)d_in[14]; const float* b1 = (const float*)d_in[15];
    const float* W2 = (const float*)d_in[16]; const float* b2 = (const float*)d_in[17];
    const float* ln2g = (const float*)d_in[18]; const float* ln2b = (const float*)d_in[19];

    const long long MBN = (long long)Bc * Nc;        // 16384
    const long long KC  = XDc + (long long)Nc * EDc; // 1398 (unpadded, f32 source)
    const long long BH  = (long long)Bc * NHc;       // 512
    const long long NN  = (long long)Nc * Nc;        // 65536

    // ---- deterministic workspace carve (bump allocator, 256B aligned) ----
    char* w = (char*)d_ws;
    size_t off = 0;
    auto carve = [&](size_t bytes) -> char* {
        char* p = w + off;
        off += (bytes + 255) & ~(size_t)255;
        return p;
    };
    float* scoresF = (float*)carve((size_t)BH * NN * 4);   // also hosts Gram & cat
    bf16*  attnB   = (bf16*) carve((size_t)BH * NN * 2);
    float* Db      = (float*)carve((size_t)Bc * NN * 4);
    float* zF      = (float*)carve((size_t)MBN * Hc * 4);
    bf16*  zB      = (bf16*) carve((size_t)MBN * Hc * 2);
    bf16*  qB      = (bf16*) carve((size_t)MBN * Hc * 2);
    bf16*  kB      = (bf16*) carve((size_t)MBN * Hc * 2);
    bf16*  vB      = (bf16*) carve((size_t)MBN * Hc * 2);
    bf16*  yB      = (bf16*) carve((size_t)MBN * Hc * 2);
    bf16*  x1B     = (bf16*) carve((size_t)MBN * Hc * 2);
    float* tmpF    = (float*)carve((size_t)MBN * Hc * 4);
    float* x1F     = (float*)carve((size_t)MBN * Hc * 4);
    bf16*  ffn1B   = (bf16*) carve((size_t)MBN * DFFc * 2);
    bf16*  wembB   = (bf16*) carve((size_t)KCP * Hc * 2);
    bf16*  wqB     = (bf16*) carve((size_t)DEPTHc * Hc * Hc * 2);
    bf16*  wkB     = (bf16*) carve((size_t)DEPTHc * Hc * Hc * 2);
    bf16*  wvB     = (bf16*) carve((size_t)DEPTHc * Hc * Hc * 2);
    bf16*  woB     = (bf16*) carve((size_t)DEPTHc * Hc * Hc * 2);
    bf16*  w1B     = (bf16*) carve((size_t)DEPTHc * Hc * DFFc * 2);
    bf16*  w2B     = (bf16*) carve((size_t)DEPTHc * DFFc * Hc * 2);
    bf16*  catB    = (bf16*)scoresF;   // alias: cat lives only before layer loop

    // ---- weight conversions f32 -> bf16 ----
    auto f2bf = [&](const float* in, bf16* out, long long n) {
        f2bf_kernel<<<dim3((unsigned)((n + 255) / 256)), dim3(256), 0, stream>>>(in, out, n);
    };
    f2bf(W_emb, wembB, KC * Hc);   // W_emb has 1398 rows; GEMM K=1398 handles tail
    f2bf(Wq, wqB, (long long)DEPTHc * Hc * Hc);
    f2bf(Wk, wkB, (long long)DEPTHc * Hc * Hc);
    f2bf(Wv, wvB, (long long)DEPTHc * Hc * Hc);
    f2bf(Wo, woB, (long long)DEPTHc * Hc * Hc);
    f2bf(W1, w1B, (long long)DEPTHc * Hc * DFFc);
    f2bf(W2, w2B, (long long)DEPTHc * DFFc * Hc);

    // ---- embed: z = cat @ W_emb  (A padded to K=1408; B keeps K=1398 w/ tail) ----
    build_cat_kernel<<<dim3((unsigned)MBN), dim3(256), 0, stream>>>(x, e, catB);
    launch_gemm<false, false>(stream, catB, wembB, nullptr, nullptr, zF, zB,
        (int)MBN, Hc, (int)KC, KCP, Hc, Hc, 0,
        0, 0, 0, 0, 0, 0, 0, 0, 1, 1, 1.0f);

    const float invSqrtDH = 1.0f / sqrtf((float)DHc);

    for (int l = 0; l < DEPTHc; ++l) {
        // Gram: G[b] = zB[b] @ zB[b]^T  (into scores buffer front)
        launch_gemm<true, false>(stream, zB, zB, nullptr, nullptr, scoresF, nullptr,
            Nc, Nc, Hc, Hc, Hc, Nc, 0,
            (long long)Nc * Hc, 0, (long long)Nc * Hc, 0, NN, 0, 0, 0,
            Bc, 1, 1.0f);
        // Distance bias D[b,i,j]
        dbias_kernel<<<dim3((unsigned)(Bc * Nc)), dim3(256), 0, stream>>>(scoresF, mask, Db);

        // QKV projections
        launch_gemm<false, false>(stream, zB, wqB + (long long)l * Hc * Hc, bq + (long long)l * Hc,
            nullptr, nullptr, qB, (int)MBN, Hc, Hc, Hc, Hc, Hc, 0,
            0, 0, 0, 0, 0, 0, 0, 0, 1, 1, 1.0f);
        launch_gemm<false, false>(stream, zB, wkB + (long long)l * Hc * Hc, bk + (long long)l * Hc,
            nullptr, nullptr, kB, (int)MBN, Hc, Hc, Hc, Hc, Hc, 0,
            0, 0, 0, 0, 0, 0, 0, 0, 1, 1, 1.0f);
        launch_gemm<false, false>(stream, zB, wvB + (long long)l * Hc * Hc, bv + (long long)l * Hc,
            nullptr, nullptr, vB, (int)MBN, Hc, Hc, Hc, Hc, Hc, 0,
            0, 0, 0, 0, 0, 0, 0, 0, 1, 1, 1.0f);

        // scores[b,h] = q_bh @ k_bh^T / sqrt(DH) + D[b]   (512 batched)
        launch_gemm<true, false>(stream, qB, kB, nullptr, Db, scoresF, nullptr,
            Nc, Nc, DHc, Hc, Hc, Nc, Nc,
            (long long)Nc * Hc, DHc, (long long)Nc * Hc, DHc,
            (long long)NHc * NN, NN, NN, 0,
            (int)BH, NHc, invSqrtDH);

        // softmax rows -> bf16 probs
        softmax_kernel<<<dim3((unsigned)(BH * Nc)), dim3(256), 0, stream>>>(scoresF, attnB);

        // y[b,h] = attn_bh @ v_bh   (512 batched; writes packed [B,N,H] bf16)
        launch_gemm<false, false>(stream, attnB, vB, nullptr, nullptr, nullptr, yB,
            Nc, DHc, Nc, Nc, Hc, Hc, 0,
            (long long)NHc * NN, NN, (long long)Nc * Hc, DHc,
            (long long)Nc * Hc, DHc, 0, 0,
            (int)BH, NHc, 1.0f);

        // output projection
        launch_gemm<false, false>(stream, yB, woB + (long long)l * Hc * Hc, bo + (long long)l * Hc,
            nullptr, tmpF, nullptr, (int)MBN, Hc, Hc, Hc, Hc, Hc, 0,
            0, 0, 0, 0, 0, 0, 0, 0, 1, 1, 1.0f);

        // x1 = LN(z + y)
        addnorm_kernel<<<dim3((unsigned)MBN), dim3(128), 0, stream>>>(
            zF, tmpF, ln1g + (long long)l * Hc, ln1b + (long long)l * Hc, x1F, x1B);

        // FFN1: relu(x1 @ W1 + b1) -> bf16
        launch_gemm<false, true>(stream, x1B, w1B + (long long)l * Hc * DFFc, b1 + (long long)l * DFFc,
            nullptr, nullptr, ffn1B, (int)MBN, DFFc, Hc, Hc, DFFc, DFFc, 0,
            0, 0, 0, 0, 0, 0, 0, 0, 1, 1, 1.0f);

        // FFN2: @ W2 + b2 -> f32
        launch_gemm<false, false>(stream, ffn1B, w2B + (long long)l * DFFc * Hc, b2 + (long long)l * Hc,
            nullptr, tmpF, nullptr, (int)MBN, Hc, DFFc, DFFc, Hc, Hc, 0,
            0, 0, 0, 0, 0, 0, 0, 0, 1, 1, 1.0f);

        // z = LN(x1 + ffn2)
        addnorm_kernel<<<dim3((unsigned)MBN), dim3(128), 0, stream>>>(
            x1F, tmpF, ln2g + (long long)l * Hc, ln2b + (long long)l * Hc, zF, zB);
    }

    // ---- masked outputs: [X | E | z] ----
    float* outX = (float*)d_out;
    float* outE = outX + (long long)Bc * Nc * XDc;
    float* outZ = outE + (long long)Bc * Nc * Nc * EDc;
    {
        long long tX = (long long)Bc * Nc * XDc;
        mask_rows_kernel<<<dim3((unsigned)((tX + 255) / 256)), dim3(256), 0, stream>>>(
            x, mask, outX, XDc, tX);
        long long tE = (long long)Bc * Nc * Nc * EDc;
        mask_e_kernel<<<dim3((unsigned)((tE + 255) / 256)), dim3(256), 0, stream>>>(
            e, mask, outE, tE);
        long long tZ = MBN * Hc;
        mask_rows_kernel<<<dim3((unsigned)((tZ + 255) / 256)), dim3(256), 0, stream>>>(
            zF, mask, outZ, Hc, tZ);
    }
}
